// PointTransformerBlock_76596446757372
// MI455X (gfx1250) — matrix operand order repaired and checked by reading, not compiled, not run on hardware
//
#include <hip/hip_runtime.h>
#include <hip/hip_bf16.h>

#define BB 4
#define NN 4096
#define CC 256
#define TT 256
#define KK 16

typedef __attribute__((ext_vector_type(16))) _Float16 v16h;
typedef __attribute__((ext_vector_type(8)))  float    v8f;

union Frag16 {
    v16h v;
    unsigned int u[8];
};

// ---------------------------------------------------------------------------
// Wave-level dual-M-tile GEMM: two 16x256 A tiles (LDS f16, rows 0-15 / 16-31)
// times W (pre-transposed f16 Wt[n*256+k]) -> four 16x16 f32 C tiles covering
// cols [n0,n0+32). Each B fragment feeds 2 WMMAs (B reuse across the 2 points).
// A and B fragments both use contiguous k = 16*half + s (16-bit A 16x32, B 32x16).
// ---------------------------------------------------------------------------
__device__ __forceinline__ void wave_gemm2_f16(
    const _Float16* As,                     // LDS [32][256]
    const _Float16* __restrict__ Wt,        // global transposed f16 [256][256]
    int lane, int n0, v8f& c00, v8f& c01, v8f& c10, v8f& c11)
{
    const int lm   = lane & 15;
    const int half = lane >> 4;
    c00 = (v8f){0.f,0.f,0.f,0.f,0.f,0.f,0.f,0.f};
    c01 = c00; c10 = c00; c11 = c00;
#pragma unroll
    for (int kk = 0; kk < 256; kk += 32) {
        Frag16 a0, a1;
#pragma unroll
        for (int j = 0; j < 8; ++j) {
            int k = kk + half * 16 + (j << 1);
            a0.u[j] = *(const unsigned int*)(As + lm * 256 + k);
            a1.u[j] = *(const unsigned int*)(As + (16 + lm) * 256 + k);
        }
        v16h b0 = *(const v16h*)(Wt + (size_t)(n0 + lm)      * 256 + kk + half * 16);
        v16h b1 = *(const v16h*)(Wt + (size_t)(n0 + 16 + lm) * 256 + kk + half * 16);
        c00 = __builtin_amdgcn_wmma_f32_16x16x32_f16(false, a0.v, false, b0,
                                                     (short)0, c00, false, false);
        c01 = __builtin_amdgcn_wmma_f32_16x16x32_f16(false, a0.v, false, b1,
                                                     (short)0, c01, false, false);
        c10 = __builtin_amdgcn_wmma_f32_16x16x32_f16(false, a1.v, false, b0,
                                                     (short)0, c10, false, false);
        c11 = __builtin_amdgcn_wmma_f32_16x16x32_f16(false, a1.v, false, b1,
                                                     (short)0, c11, false, false);
    }
}

// Single-M-tile version for the row-panel GEMM kernel.
__device__ __forceinline__ void wave_gemm_f16(
    const _Float16* As, const _Float16* __restrict__ Wt,
    int lane, int n0, v8f& c0, v8f& c1)
{
    const int lm   = lane & 15;
    const int half = lane >> 4;
    c0 = (v8f){0.f,0.f,0.f,0.f,0.f,0.f,0.f,0.f};
    c1 = c0;
#pragma unroll
    for (int kk = 0; kk < 256; kk += 32) {
        Frag16 a;
#pragma unroll
        for (int j = 0; j < 8; ++j) {
            int k = kk + half * 16 + (j << 1);
            a.u[j] = *(const unsigned int*)(As + lm * 256 + k);
        }
        v16h b0 = *(const v16h*)(Wt + (size_t)(n0 + lm)      * 256 + kk + half * 16);
        v16h b1 = *(const v16h*)(Wt + (size_t)(n0 + 16 + lm) * 256 + kk + half * 16);
        c0 = __builtin_amdgcn_wmma_f32_16x16x32_f16(false, a.v, false, b0,
                                                    (short)0, c0, false, false);
        c1 = __builtin_amdgcn_wmma_f32_16x16x32_f16(false, a.v, false, b1,
                                                    (short)0, c1, false, false);
    }
}

// ---------------------------------------------------------------------------
// Weight transpose + f32 -> f16 convert: dst[n*256+k] = src[k*256+n]
// ---------------------------------------------------------------------------
__global__ void __launch_bounds__(256)
w_transpose_kernel(const float* __restrict__ src, _Float16* __restrict__ dst)
{
    int n = blockIdx.x;
    int k = threadIdx.x;
    dst[(size_t)n * 256 + k] = (_Float16)src[(size_t)k * 256 + n];
}

// ---------------------------------------------------------------------------
// KNN: per query point, top-16 smallest (||a||^2+||b||^2-2ab), stable ties.
// ---------------------------------------------------------------------------
__global__ void __launch_bounds__(256)
knn_kernel(const float* __restrict__ pos, int* __restrict__ knn)
{
    __shared__ float cpos[256 * 3];
    const int tid = threadIdx.x;
    const int gp  = blockIdx.x * 256 + tid;   // blocks never straddle batches
    const int b   = gp >> 12;                 // N = 4096
    const float* bpos = pos + (size_t)b * NN * 3;

    const float qx = pos[(size_t)gp * 3 + 0];
    const float qy = pos[(size_t)gp * 3 + 1];
    const float qz = pos[(size_t)gp * 3 + 2];
    const float qn2 = qx * qx + qy * qy + qz * qz;

    float bd[16];
    int   bi[16];
#pragma unroll
    for (int s = 0; s < 16; ++s) { bd[s] = 3.4e38f; bi[s] = 0; }

    for (int tile = 0; tile < NN; tile += 256) {
        __syncthreads();
#pragma unroll
        for (int d = 0; d < 3; ++d)
            cpos[tid * 3 + d] = bpos[(size_t)(tile + tid) * 3 + d];
        __syncthreads();
        for (int j = 0; j < 256; ++j) {
            float cx = cpos[j * 3 + 0];
            float cy = cpos[j * 3 + 1];
            float cz = cpos[j * 3 + 2];
            float cn2 = cx * cx + cy * cy + cz * cz;
            float d = qn2 + cn2 - 2.f * (qx * cx + qy * cy + qz * cz);
            if (d < bd[15]) {
                bd[15] = d; bi[15] = tile + j;
#pragma unroll
                for (int s = 15; s > 0; --s) {
                    if (bd[s] < bd[s - 1]) {   // strict: stable tie order
                        float td = bd[s]; bd[s] = bd[s - 1]; bd[s - 1] = td;
                        int   ti = bi[s]; bi[s] = bi[s - 1]; bi[s - 1] = ti;
                    }
                }
            }
        }
    }
#pragma unroll
    for (int s = 0; s < 16; ++s) knn[(size_t)gp * 16 + s] = bi[s];
}

// ---------------------------------------------------------------------------
// Generic [M,256]x[256,256] WMMA GEMM (+optional bias, +optional residual).
// Block = 16 rows x 256 cols, 8 waves.
// ---------------------------------------------------------------------------
template <bool A_HALF, bool OUT_HALF>
__global__ void __launch_bounds__(256)
gemm256_kernel(const void* __restrict__ Ain, const _Float16* __restrict__ Wt,
               const float* __restrict__ bias, const float* __restrict__ resid,
               void* __restrict__ Out)
{
    __shared__ _Float16 As[16 * 256];
    const int tid = threadIdx.x;
    const int m0  = blockIdx.x * 16;

    if (A_HALF) {
        const _Float16* A = (const _Float16*)Ain;
#pragma unroll
        for (int i = 0; i < 16; ++i)
            As[i * 256 + tid] = A[(size_t)(m0 + i) * 256 + tid];
    } else {
        const float* A = (const float*)Ain;
#pragma unroll
        for (int i = 0; i < 16; ++i)
            As[i * 256 + tid] = (_Float16)A[(size_t)(m0 + i) * 256 + tid];
    }
    __syncthreads();

    const int lane = tid & 31;
    const int wave = tid >> 5;
    const int n0   = wave * 32;
    v8f c0, c1;
    wave_gemm_f16(As, Wt, lane, n0, c0, c1);

    const int lm = lane & 15, half = lane >> 4;
#pragma unroll
    for (int r = 0; r < 8; ++r) {
        int row  = m0 + half * 8 + r;
        int col0 = n0 + lm, col1 = n0 + 16 + lm;
        float v0 = c0[r] + (bias ? bias[col0] : 0.f)
                 + (resid ? resid[(size_t)row * 256 + col0] : 0.f);
        float v1 = c1[r] + (bias ? bias[col1] : 0.f)
                 + (resid ? resid[(size_t)row * 256 + col1] : 0.f);
        if (OUT_HALF) {
            ((_Float16*)Out)[(size_t)row * 256 + col0] = (_Float16)v0;
            ((_Float16*)Out)[(size_t)row * 256 + col1] = (_Float16)v1;
        } else {
            ((float*)Out)[(size_t)row * 256 + col0] = v0;
            ((float*)Out)[(size_t)row * 256 + col1] = v1;
        }
    }
}

// ---------------------------------------------------------------------------
// Per-column softmax over K=16 rows of one C tile, entirely in registers.
// Column col: rows half*8+r live in this lane; the other 8 rows live in
// lane^16 (same column). One shfl_xor(16) completes the M reduction (wave32).
// Writes attn to global and accumulates res_pre[col] = sum_k a*(v+pe).
// ---------------------------------------------------------------------------
__device__ __forceinline__ void softmax_col(
    const v8f& c, int gp, int b, int col, int half, int p,
    const float* __restrict__ g2_b, const float* __restrict__ hv,
    const _Float16* peH, const int* nidx,
    float* __restrict__ out_attn, _Float16* __restrict__ resp16)
{
    const float scale = 0.0625f;            // 1/sqrt(256)
    const float gb = g2_b[col];
    float a[8];
    float mx = -3.4e38f;
#pragma unroll
    for (int r = 0; r < 8; ++r) { a[r] = (c[r] + gb) * scale; mx = fmaxf(mx, a[r]); }
    mx = fmaxf(mx, __shfl_xor(mx, 16, 32));
    float s = 0.f;
#pragma unroll
    for (int r = 0; r < 8; ++r) { a[r] = __expf(a[r] - mx); s += a[r]; }
    s += __shfl_xor(s, 16, 32);
    const float inv = 1.f / s;
    float acc = 0.f;
#pragma unroll
    for (int r = 0; r < 8; ++r) {
        int jj = half * 8 + r;              // neighbor index within this point
        float av = a[r] * inv;
        out_attn[((size_t)gp * 16 + jj) * 256 + col] = av;
        float vv = hv[((size_t)b * NN + nidx[p * 16 + jj]) * 256 + col]
                 + (float)peH[(p * 16 + jj) * 256 + col];
        acc += av * vv;
    }
    acc += __shfl_xor(acc, 16, 32);
    if (half == 0) resp16[(size_t)gp * 256 + col] = (_Float16)acc;
}

// ---------------------------------------------------------------------------
// Fused attention: one block per 2 points (2 M-tiles share every B fragment).
// ---------------------------------------------------------------------------
__global__ void __launch_bounds__(256)
pt_attn_kernel(const float* __restrict__ pos, const int* __restrict__ knn,
               const float* __restrict__ qbuf, const float* __restrict__ hk,
               const float* __restrict__ hv,
               const float* __restrict__ d1_w, const float* __restrict__ d1_b,
               const _Float16* __restrict__ d2t, const float* __restrict__ d2_b,
               const _Float16* __restrict__ g1t, const float* __restrict__ g1_b,
               const _Float16* __restrict__ g2t, const float* __restrict__ g2_b,
               float* __restrict__ out_attn, _Float16* __restrict__ resp16)
{
    __shared__ _Float16 As[32 * 256];    // 16 KB (GEMM A operand, reused)
    __shared__ _Float16 Tb[32 * 256];    // 16 KB (g1 output)
    __shared__ _Float16 peH[32 * 256];   // 16 KB (pos_enc)
    __shared__ float    qn[2 * 256];
    __shared__ float    rel[32 * 4];
    __shared__ int      nidx[32];

    const int tid = threadIdx.x;
    const int gp0 = blockIdx.x * 2;      // first of the 2 points
    const int b   = gp0 >> 12;

    qn[tid]       = qbuf[(size_t)gp0 * 256 + tid];
    qn[256 + tid] = qbuf[(size_t)(gp0 + 1) * 256 + tid];
    if (tid < 32) nidx[tid] = knn[(size_t)gp0 * 16 + tid];  // both points, contiguous
    __syncthreads();
    if (tid < 96) {
        int j = tid / 3, d = tid - j * 3;
        int p = j >> 4;
        rel[j * 4 + d] = pos[(size_t)(gp0 + p) * 3 + d]
                       - pos[((size_t)b * NN + nidx[j]) * 3 + d];
    }
    __syncthreads();

    // stage 1: As = f16(relu(rel @ d1_w + d1_b))            [32 x 256]
    {
        float w0 = d1_w[tid], w1 = d1_w[256 + tid], w2 = d1_w[512 + tid];
        float bb = d1_b[tid];
#pragma unroll
        for (int j = 0; j < 32; ++j) {
            float v = rel[j * 4 + 0] * w0 + rel[j * 4 + 1] * w1
                    + rel[j * 4 + 2] * w2 + bb;
            As[j * 256 + tid] = (_Float16)(v > 0.f ? v : 0.f);
        }
    }
    __syncthreads();

    const int lane = tid & 31, wave = tid >> 5;
    const int lm = lane & 15, half = lane >> 4;
    const int n0 = wave * 32;
    const int col0 = n0 + lm, col1 = n0 + 16 + lm;

    // stage 2: peH = As @ d2_w + d2_b   (WMMA, dual M-tile)
    {
        v8f c00, c01, c10, c11;
        wave_gemm2_f16(As, d2t, lane, n0, c00, c01, c10, c11);
        float b0 = d2_b[col0], b1 = d2_b[col1];
#pragma unroll
        for (int r = 0; r < 8; ++r) {
            int row = half * 8 + r;
            peH[row * 256 + col0]        = (_Float16)(c00[r] + b0);
            peH[row * 256 + col1]        = (_Float16)(c01[r] + b1);
            peH[(16 + row) * 256 + col0] = (_Float16)(c10[r] + b0);
            peH[(16 + row) * 256 + col1] = (_Float16)(c11[r] + b1);
        }
    }
    __syncthreads();

    // stage 3: As = f16(q - k_gathered + pe)
#pragma unroll
    for (int j = 0; j < 32; ++j) {
        int p = j >> 4;
        float kv = hk[((size_t)b * NN + nidx[j]) * 256 + tid];
        As[j * 256 + tid] =
            (_Float16)(qn[p * 256 + tid] - kv + (float)peH[j * 256 + tid]);
    }
    __syncthreads();

    // stage 4: Tb = f16(relu(As @ g1_w + g1_b))   (WMMA, dual M-tile)
    {
        v8f c00, c01, c10, c11;
        wave_gemm2_f16(As, g1t, lane, n0, c00, c01, c10, c11);
        float b0 = g1_b[col0], b1 = g1_b[col1];
#pragma unroll
        for (int r = 0; r < 8; ++r) {
            int row = half * 8 + r;
            float v00 = c00[r] + b0, v01 = c01[r] + b1;
            float v10 = c10[r] + b0, v11 = c11[r] + b1;
            Tb[row * 256 + col0]        = (_Float16)(v00 > 0.f ? v00 : 0.f);
            Tb[row * 256 + col1]        = (_Float16)(v01 > 0.f ? v01 : 0.f);
            Tb[(16 + row) * 256 + col0] = (_Float16)(v10 > 0.f ? v10 : 0.f);
            Tb[(16 + row) * 256 + col1] = (_Float16)(v11 > 0.f ? v11 : 0.f);
        }
    }
    __syncthreads();

    // stage 5: logits = Tb @ g2_w + g2_b (WMMA), softmax over K in registers,
    // write attn + accumulate res_pre (f16 to workspace for the fc2 GEMM).
    {
        v8f c00, c01, c10, c11;
        wave_gemm2_f16(Tb, g2t, lane, n0, c00, c01, c10, c11);
        softmax_col(c00, gp0,     b, col0, half, 0, g2_b, hv, peH, nidx, out_attn, resp16);
        softmax_col(c01, gp0,     b, col1, half, 0, g2_b, hv, peH, nidx, out_attn, resp16);
        softmax_col(c10, gp0 + 1, b, col0, half, 1, g2_b, hv, peH, nidx, out_attn, resp16);
        softmax_col(c11, gp0 + 1, b, col1, half, 1, g2_b, hv, peH, nidx, out_attn, resp16);
    }
}

// ---------------------------------------------------------------------------
extern "C" void kernel_launch(void* const* d_in, const int* in_sizes, int n_in,
                              void* d_out, int out_size, void* d_ws, size_t ws_size,
                              hipStream_t stream)
{
    const float* x     = (const float*)d_in[0];
    const float* pos   = (const float*)d_in[1];
    const float* fc1_w = (const float*)d_in[2];
    const float* fc1_b = (const float*)d_in[3];
    const float* fc2_w = (const float*)d_in[4];
    const float* fc2_b = (const float*)d_in[5];
    const float* d1_w  = (const float*)d_in[6];
    const float* d1_b  = (const float*)d_in[7];
    const float* d2_w  = (const float*)d_in[8];
    const float* d2_b  = (const float*)d_in[9];
    const float* g1_w  = (const float*)d_in[10];
    const float* g1_b  = (const float*)d_in[11];
    const float* g2_w  = (const float*)d_in[12];
    const float* g2_b  = (const float*)d_in[13];
    const float* wq    = (const float*)d_in[14];
    const float* wk    = (const float*)d_in[15];
    const float* wv    = (const float*)d_in[16];

    const size_t M = (size_t)BB * NN;            // 16384 rows

    // workspace layout
    char* w = (char*)d_ws;
    int*      knn    = (int*)w;       w += M * KK * sizeof(int);
    _Float16* h16    = (_Float16*)w;  w += M * TT * sizeof(_Float16);
    _Float16* resp16 = (_Float16*)w;  w += M * TT * sizeof(_Float16);
    float*    q      = (float*)w;     w += M * TT * sizeof(float);
    float*    hk     = (float*)w;     w += M * TT * sizeof(float);
    float*    hv     = (float*)w;     w += M * TT * sizeof(float);
    _Float16* fc1t   = (_Float16*)w;  w += 256 * 256 * sizeof(_Float16);
    _Float16* fc2t   = (_Float16*)w;  w += 256 * 256 * sizeof(_Float16);
    _Float16* wqt    = (_Float16*)w;  w += 256 * 256 * sizeof(_Float16);
    _Float16* wkt    = (_Float16*)w;  w += 256 * 256 * sizeof(_Float16);
    _Float16* wvt    = (_Float16*)w;  w += 256 * 256 * sizeof(_Float16);
    _Float16* d2t    = (_Float16*)w;  w += 256 * 256 * sizeof(_Float16);
    _Float16* g1t    = (_Float16*)w;  w += 256 * 256 * sizeof(_Float16);
    _Float16* g2t    = (_Float16*)w;  w += 256 * 256 * sizeof(_Float16);

    float* out_res  = (float*)d_out;                 // [B,N,C]
    float* out_attn = out_res + M * CC;              // [B,N,K,T]

    // 1. pre-transpose/convert the 8 WMMA weights to f16 [N][K]
    const float*   wsrc[8] = { fc1_w, fc2_w, wq,  wk,  wv,  d2_w, g1_w, g2_w };
    _Float16*      wdst[8] = { fc1t,  fc2t,  wqt, wkt, wvt, d2t,  g1t,  g2t  };
    for (int i = 0; i < 8; ++i)
        w_transpose_kernel<<<256, 256, 0, stream>>>(wsrc[i], wdst[i]);

    // 2. KNN
    knn_kernel<<<M / 256, 256, 0, stream>>>(pos, knn);

    // 3. h = x @ fc1_w + fc1_b   (f32 in, f16 out)
    gemm256_kernel<false, true><<<M / 16, 256, 0, stream>>>(x, fc1t, fc1_b, nullptr, h16);
    // 4. q / hk / hv = h @ {wq,wk,wv}   (f16 in, f32 out, no bias)
    gemm256_kernel<true, false><<<M / 16, 256, 0, stream>>>(h16, wqt, nullptr, nullptr, q);
    gemm256_kernel<true, false><<<M / 16, 256, 0, stream>>>(h16, wkt, nullptr, nullptr, hk);
    gemm256_kernel<true, false><<<M / 16, 256, 0, stream>>>(h16, wvt, nullptr, nullptr, hv);

    // 5. fused per-point attention (2 points/block), emits attn + res_pre(f16)
    pt_attn_kernel<<<M / 2, 256, 0, stream>>>(pos, knn, q, hk, hv,
                                              d1_w, d1_b, d2t, d2_b,
                                              g1t, g1_b, g2t, g2_b,
                                              out_attn, resp16);

    // 6. res = res_pre @ fc2_w + fc2_b + x   (WMMA with fused residual)
    gemm256_kernel<true, false><<<M / 16, 256, 0, stream>>>(resp16, fc2t, fc2_b, x, out_res);
}
